// L1_EmbbedingGNN_74217034875542
// MI455X (gfx1250) — matrix operand order repaired and checked by reading, not compiled, not run on hardware
//
#include <hip/hip_runtime.h>
#include <hip/hip_bf16.h>
#include <math.h>

// ---------------- problem constants (from reference) ----------------
#define NI_  100000
#define NO_  300000
#define NM_  50000
#define NR_  20000
#define EIA_ 100000
#define EOA_ 300000
#define ENM_ 600000
#define ENR_ 600000
#define EP_  300000

typedef __bf16 bf16_t;
typedef bf16_t v16bf __attribute__((ext_vector_type(16)));
typedef float  v8f   __attribute__((ext_vector_type(8)));
typedef float  v4f   __attribute__((ext_vector_type(4)));

// ---------------- small device helpers ----------------
__device__ __forceinline__ void atomAddF(float* p, float v) {
  __hip_atomic_fetch_add(p, v, __ATOMIC_RELAXED, __HIP_MEMORY_SCOPE_AGENT);
}
// order-preserving float -> uint key (for atomicMax over signed floats)
__device__ __forceinline__ unsigned fkey(float f) {
  unsigned u = __float_as_uint(f);
  return (u & 0x80000000u) ? ~u : (u | 0x80000000u);
}
__device__ __forceinline__ float fdec(unsigned k) {
  unsigned u = (k & 0x80000000u) ? (k & 0x7fffffffu) : ~k;
  return __uint_as_float(u);
}

// ---------------- WMMA GEMM: out = act(gather(A) @ W + b) ----------------
// One wave owns a 16x32 output tile (two 16x16 WMMA accumulators sharing one
// A fragment). The k-loop has a wave-UNIFORM fast/slow split:
//  * fast path (kb+32<=K && n0+32<=N): all chunk loads from two base pointers
//    with immediate offsets -> clause-batched b128 loads, single wait, no mask
//  * slow path (tails, N=8): clamped addresses + per-lane B mask (branchless).
// Garbage in clamped A rows / B cols lands only in rows/cols the store guard
// discards; k-tail garbage is killed by zeroing the B side.
__global__ void wmma_gemm_kernel(const float* __restrict__ A,
                                 const int* __restrict__ gidx,
                                 int M, int K,
                                 const float* __restrict__ W,
                                 const float* __restrict__ bias,
                                 int N,
                                 float* __restrict__ out,
                                 int ostride, int ocol, int act)
{
  const int lane   = threadIdx.x & 31;
  const int wave   = threadIdx.x >> 5;
  const int nPairs = (N + 31) >> 5;               // 32-column pairs of n-tiles
  const int pair   = blockIdx.y * (blockDim.x >> 5) + wave;
  if (pair >= nPairs) return;        // wave-uniform exit: EXEC stays all-ones
  const int mtile = blockIdx.x;
  const int half  = lane >> 4;
  const int l15   = lane & 15;

  int mrow = mtile * 16 + l15;
  if (mrow > M - 1) mrow = M - 1;    // clamp: garbage rows never stored
  const long asrc = gidx ? (long)gidx[mrow] : (long)mrow;
  const float* __restrict__ Arow = A + asrc * (long)K;

  const int  n0    = pair << 5;      // first column of the 32-wide tile
  const bool fullN = (n0 + 32 <= N); // uniform across the wave
  v8f c0 = {0.f, 0.f, 0.f, 0.f, 0.f, 0.f, 0.f, 0.f};
  v8f c1 = {0.f, 0.f, 0.f, 0.f, 0.f, 0.f, 0.f, 0.f};

  for (int kb = 0; kb < K; kb += 32) {
    v16bf a, b0, b1;
    const int krow = kb + l15 + 16 * half;   // unused; this lane loads B k-rows kb+8*half+{0..7,16..23}

    if ((kb + 32 <= K) & fullN) {
      // ---------- fast path: immediate-offset chunk loads, no masking ----
      const float* __restrict__ Ab = Arow + kb + 8 * half;
      v4f a0 = *(const v4f*)(Ab + 0);
      v4f a1 = *(const v4f*)(Ab + 4);
      v4f a2 = *(const v4f*)(Ab + 16);
      v4f a3 = *(const v4f*)(Ab + 20);
      const float* __restrict__ Bb = W + (long)(kb + 8 * half) * N + n0 + l15;
#pragma unroll
      for (int e = 0; e < 4; ++e) {
        a[e]       = (bf16_t)a0[e];
        a[e + 4]   = (bf16_t)a1[e];
        a[e + 8]   = (bf16_t)a2[e];
        a[e + 12]  = (bf16_t)a3[e];
        b0[e]      = (bf16_t)Bb[(long)e * N];
        b0[e + 4]  = (bf16_t)Bb[(long)(e + 4) * N];
        b0[e + 8]  = (bf16_t)Bb[(long)(e + 16) * N];
        b0[e + 12] = (bf16_t)Bb[(long)(e + 20) * N];
        b1[e]      = (bf16_t)Bb[(long)e * N + 16];
        b1[e + 4]  = (bf16_t)Bb[(long)(e + 4) * N + 16];
        b1[e + 8]  = (bf16_t)Bb[(long)(e + 16) * N + 16];
        b1[e + 12] = (bf16_t)Bb[(long)(e + 20) * N + 16];
      }
    } else {
      // ---------- slow path: clamped addresses + per-lane B mask ----------
      int p1 = kb + 8 * half;       if (p1 > K - 8) p1 = K - 8;
      int p2 = kb + 16 + 8 * half;  if (p2 > K - 8) p2 = K - 8;
      v4f a0 = *(const v4f*)(Arow + p1);
      v4f a1 = *(const v4f*)(Arow + p1 + 4);
      v4f a2 = *(const v4f*)(Arow + p2);
      v4f a3 = *(const v4f*)(Arow + p2 + 4);
#pragma unroll
      for (int e = 0; e < 4; ++e) {
        a[e]      = (bf16_t)a0[e];
        a[e + 4]  = (bf16_t)a1[e];
        a[e + 8]  = (bf16_t)a2[e];
        a[e + 12] = (bf16_t)a3[e];
      }
      int nc0 = n0 + l15;       if (nc0 > N - 1) nc0 = N - 1;
      int nc1 = n0 + 16 + l15;  if (nc1 > N - 1) nc1 = N - 1;
#pragma unroll
      for (int s = 0; s < 16; ++s) {
        const int   kk   = kb + 8 * half + (s < 8 ? s : s + 8);
        const float mask = (kk < K) ? 1.f : 0.f;   // zero k-tail
        const long  krc  = (kk < K) ? kk : (K - 1);
        b0[s] = (bf16_t)(W[krc * N + nc0] * mask);
        b1[s] = (bf16_t)(W[krc * N + nc1] * mask);
      }
    }
    c0 = __builtin_amdgcn_wmma_f32_16x16x32_bf16(false, a, false, b0,
                                                 (short)0, c0, false, false);
    c1 = __builtin_amdgcn_wmma_f32_16x16x32_bf16(false, a, false, b1,
                                                 (short)0, c1, false, false);
  }

  // C/D layout: lane n = l15, VGPR r holds row (r + 8*half)
  const int  nA = n0 + l15;
  const int  nB = n0 + 16 + l15;
  const bool okA = (nA < N);
  const bool okB = (nB < N);
  const float bvA = bias ? bias[okA ? nA : (N - 1)] : 0.f;
  const float bvB = bias ? bias[okB ? nB : (N - 1)] : 0.f;
#pragma unroll
  for (int r = 0; r < 8; ++r) {
    const int row = mtile * 16 + r + 8 * half;
    if (row < M) {
      float* orow = out + (long)row * ostride + ocol;
      if (okA) {
        float v = c0[r] + bvA;
        if (act == 1) v = fmaxf(v, 0.f);
        orow[nA] = v;
      }
      if (okB) {
        float v = c1[r] + bvB;
        if (act == 1) v = fmaxf(v, 0.f);
        orow[nB] = v;
      }
    }
  }
}

// ---------------- segment-sum scatter-add ----------------
// out[didx[e]*dim + d] += src[sidx[e]*dim + d]
__global__ void segsum_kernel(const float* __restrict__ src,
                              const int* __restrict__ sidx,
                              const int* __restrict__ didx,
                              long E, int dim,
                              float* __restrict__ out)
{
  long i = (long)blockIdx.x * blockDim.x + threadIdx.x;
  if (i >= E * dim) return;
  int e = (int)(i / dim);
  int d = (int)(i % dim);
  float v = src[(long)sidx[e] * dim + d];
  atomAddF(&out[(long)didx[e] * dim + d], v);
}

// ---------------- material attention pieces ----------------
// cat32[e] = [operations[eop[e]][0:24], nfm_attr[e][0:8]]
__global__ void cat32_kernel(const float* __restrict__ ops,
                             const float* __restrict__ attr,
                             const int* __restrict__ eop,
                             long E, float* __restrict__ out)
{
  long i = (long)blockIdx.x * blockDim.x + threadIdx.x;
  if (i >= E * 32) return;
  int e = (int)(i >> 5);
  int d = (int)(i & 31);
  out[i] = (d < 24) ? ops[(long)eop[e] * 24 + d] : attr[(long)e * 8 + (d - 24)];
}

// att_all[i] = leaky_relu( up_m[i] . (att[0:64]+att[64:128]), 0.2 )
__global__ void self_att_kernel(const float* __restrict__ up_m,
                                const float* __restrict__ att,
                                int NM, float* __restrict__ att_all)
{
  int i = blockIdx.x * blockDim.x + threadIdx.x;
  if (i >= NM) return;
  float s = 0.f;
  const float* r = up_m + (long)i * 64;
#pragma unroll 8
  for (int j = 0; j < 64; ++j) s += r[j] * (att[j] + att[64 + j]);
  att_all[i] = s > 0.f ? s : 0.2f * s;
}

// att_all[NM+e] = leaky_relu( [up_m[emat[e]], opsfeat[e]] . att, 0.2 )
__global__ void cross_att_kernel(const float* __restrict__ up_m,
                                 const float* __restrict__ opsfeat,
                                 const int* __restrict__ emat,
                                 const float* __restrict__ att,
                                 long E, int NM, float* __restrict__ att_all)
{
  long e = (long)blockIdx.x * blockDim.x + threadIdx.x;
  if (e >= E) return;
  const float* m = up_m + (long)emat[e] * 64;
  const float* o = opsfeat + e * 64;
  float s = 0.f;
#pragma unroll 8
  for (int j = 0; j < 64; ++j) s += m[j] * att[j] + o[j] * att[64 + j];
  att_all[NM + e] = s > 0.f ? s : 0.2f * s;
}

__global__ void init_scal_kernel(unsigned* mk, float* se)
{
  if (threadIdx.x == 0 && blockIdx.x == 0) { *mk = fkey(-1e30f); *se = 0.f; }
}

__global__ void reduce_max_kernel(const float* __restrict__ x, long n, unsigned* mk)
{
  __shared__ float red[256];
  long i = (long)blockIdx.x * blockDim.x + threadIdx.x;
  const long stride = (long)gridDim.x * blockDim.x;
  float m = -1e30f;
  for (; i < n; i += stride) m = fmaxf(m, x[i]);
  red[threadIdx.x] = m; __syncthreads();
  for (int s = blockDim.x >> 1; s > 0; s >>= 1) {
    if ((int)threadIdx.x < s) red[threadIdx.x] = fmaxf(red[threadIdx.x], red[threadIdx.x + s]);
    __syncthreads();
  }
  if (threadIdx.x == 0) atomicMax(mk, fkey(red[0]));
}

__global__ void reduce_sumexp_kernel(const float* __restrict__ x, long n,
                                     const unsigned* __restrict__ mk, float* se)
{
  __shared__ float red[256];
  const float mx = fdec(*mk);
  long i = (long)blockIdx.x * blockDim.x + threadIdx.x;
  const long stride = (long)gridDim.x * blockDim.x;
  float s = 0.f;
  for (; i < n; i += stride) s += __expf(x[i] - mx);
  red[threadIdx.x] = s; __syncthreads();
  for (int t = blockDim.x >> 1; t > 0; t >>= 1) {
    if ((int)threadIdx.x < t) red[threadIdx.x] += red[threadIdx.x + t];
    __syncthreads();
  }
  if (threadIdx.x == 0) atomAddF(se, red[0]);
}

// sum_ops[emat[e]] += softmax(cross_att[e]) * opsfeat[e]
__global__ void attn_segsum_kernel(const float* __restrict__ opsfeat,
                                   const float* __restrict__ att_all,
                                   const int* __restrict__ emat,
                                   long E, int NM,
                                   const unsigned* __restrict__ mk,
                                   const float* __restrict__ se,
                                   float* __restrict__ out)
{
  long i = (long)blockIdx.x * blockDim.x + threadIdx.x;
  if (i >= E * 64) return;
  int e = (int)(i >> 6);
  int d = (int)(i & 63);
  float w = __expf(att_all[NM + e] - fdec(*mk)) / (*se);
  atomAddF(&out[(long)emat[e] * 64 + d], w * opsfeat[i]);
}

// mat_emb = elu( softmax(self_att) * up_m + sum_ops )
__global__ void mat_emb_kernel(const float* __restrict__ up_m,
                               const float* __restrict__ sum_ops,
                               const float* __restrict__ att_all,
                               const unsigned* __restrict__ mk,
                               const float* __restrict__ se,
                               int NM, float* __restrict__ out)
{
  long i = (long)blockIdx.x * blockDim.x + threadIdx.x;
  if (i >= (long)NM * 64) return;
  int m = (int)(i >> 6);
  float w = __expf(att_all[m] - fdec(*mk)) / (*se);
  float v = w * up_m[i] + sum_ops[i];
  out[i] = v > 0.f ? v : (__expf(v) - 1.f);
}

// ---------------- host-side helpers ----------------
static inline int blocks1d(long n, int bs) { return (int)((n + bs - 1) / bs); }

static void gemm(hipStream_t s, const float* A, const int* gidx, int M, int K,
                 const float* W, const float* b, int N,
                 float* out, int ostride, int ocol, int act)
{
  int nPairs = (N + 31) / 32;            // each wave = one 16x32 output tile
  int WPB = nPairs < 4 ? nPairs : 4;     // waves per block
  dim3 grid((M + 15) / 16, (nPairs + WPB - 1) / WPB);
  dim3 block(32 * WPB);
  wmma_gemm_kernel<<<grid, block, 0, s>>>(A, gidx, M, K, W, b, N, out, ostride, ocol, act);
}

extern "C" void kernel_launch(void* const* d_in, const int* in_sizes, int n_in,
                              void* d_out, int out_size, void* d_ws, size_t ws_size,
                              hipStream_t stream)
{
#define F(i) ((const float*)d_in[(i)])
#define I(i) ((const int*)d_in[(i)])
  // top-level insertion order; 'params' flattened with sorted keys (jax tree order)
  const float* items      = F(0);
  const float* operations = F(1);
  const float* materials  = F(2);
  const float* resources  = F(3);
  const float* nfm_attr   = F(4);
  // item_children {W1,W2,b1,b2}
  const float *ich_W1=F(5),  *ich_W2=F(6),  *ich_b1=F(7),  *ich_b2=F(8);
  // item_combined {W1,W2,W3,b1,b2,b3}
  const float *icb_W1=F(9),  *icb_W2=F(10), *icb_W3=F(11), *icb_b1=F(12), *icb_b2=F(13), *icb_b3=F(14);
  // item_ops
  const float *iop_W1=F(15), *iop_W2=F(16), *iop_b1=F(17), *iop_b2=F(18);
  // item_parent
  const float *ipa_W1=F(19), *ipa_W2=F(20), *ipa_b1=F(21), *ipa_b2=F(22);
  // item_self
  const float *ise_W1=F(23), *ise_W2=F(24), *ise_b1=F(25), *ise_b2=F(26);
  const float *mat_Wm   = F(27);
  const float *mat_Wop  = F(28);
  const float *att_cross= F(29);
  const float *att_self = F(30);
  // op_combined
  const float *ocb_W1=F(31), *ocb_W2=F(32), *ocb_W3=F(33), *ocb_b1=F(34), *ocb_b2=F(35), *ocb_b3=F(36);
  // op_items
  const float *oit_W1=F(37), *oit_W2=F(38), *oit_b1=F(39), *oit_b2=F(40);
  // op_mat
  const float *oma_W1=F(41), *oma_W2=F(42), *oma_b1=F(43), *oma_b2=F(44);
  // op_pred
  const float *opr_W1=F(45), *opr_W2=F(46), *opr_b1=F(47), *opr_b2=F(48);
  // op_res
  const float *ore_W1=F(49), *ore_W2=F(50), *ore_b1=F(51), *ore_b2=F(52);
  // op_self
  const float *ose_W1=F(53), *ose_W2=F(54), *ose_b1=F(55), *ose_b2=F(56);
  // op_succ
  const float *osu_W1=F(57), *osu_W2=F(58), *osu_b1=F(59), *osu_b2=F(60);
  const int *parents = I(61);
  const int *related = I(62);
  const int *ia  = I(63);   // [2, EIA]
  const int *oa  = I(64);   // [2, EOA]
  const int *nfm = I(65);   // [2, ENM]
  const int *nfr = I(66);   // [2, ENR]
  const int *prc = I(67);   // [2, EP]
#undef F
#undef I

  float* item_out = (float*)d_out;
  float* op_out   = item_out + (size_t)NI_ * 64;
  float* mat_out  = op_out   + (size_t)NO_ * 64;

  // ---- bump allocator over d_ws (256B aligned -> float4 loads legal) ----
  char* wp = (char*)d_ws;
  auto alloc = [&](size_t nfl) -> float* {
    float* p = (float*)wp;
    wp += ((nfl * sizeof(float)) + 255) & ~(size_t)255;
    return p;
  };
  float* up_m      = alloc((size_t)NM_ * 64);
  float* att_all   = alloc((size_t)NM_ + ENM_);
  float* cat32     = alloc((size_t)ENM_ * 32);
  float* opsfeat   = alloc((size_t)ENM_ * 64);
  float* sum_ops   = alloc((size_t)NM_ * 64);
  unsigned* mk     = (unsigned*)alloc(64);
  float* se        = alloc(64);
  float* agg_child = alloc((size_t)NI_ * 16);
  float* agg_ops   = alloc((size_t)NI_ * 24);
  float* comb_item = alloc((size_t)NI_ * 256);
  float* hidden    = alloc((size_t)NO_ * 128);   // reused by every MLP stage
  float* h2        = alloc((size_t)NO_ * 64);
  float* agg_mat   = alloc((size_t)NO_ * 8);
  float* agg_res   = alloc((size_t)NO_ * 8);
  float* agg_pred  = alloc((size_t)NO_ * 24);
  float* agg_succ  = alloc((size_t)NO_ * 24);
  float* comb_op   = alloc((size_t)NO_ * 272);

  // ================= Phase A: item embeddings =================
  hipMemsetAsync(agg_child, 0, (size_t)NI_ * 16 * 4, stream);
  hipMemsetAsync(agg_ops,   0, (size_t)NI_ * 24 * 4, stream);
  { long n = (long)EIA_ * 16;
    segsum_kernel<<<blocks1d(n,256),256,0,stream>>>(items, ia + EIA_, ia, EIA_, 16, agg_child); }
  { long n = (long)EOA_ * 24;
    segsum_kernel<<<blocks1d(n,256),256,0,stream>>>(operations, oa + EOA_, oa, EOA_, 24, agg_ops); }

  // comb_item layout: [parent(0) | child(64) | ops(128) | self(192)]
  gemm(stream, items,     parents, NI_, 16,  ipa_W1, ipa_b1, 128, hidden, 128, 0, 1);
  gemm(stream, hidden,    nullptr, NI_, 128, ipa_W2, ipa_b2, 64,  comb_item, 256, 0,   0);
  gemm(stream, agg_child, nullptr, NI_, 16,  ich_W1, ich_b1, 128, hidden, 128, 0, 1);
  gemm(stream, hidden,    nullptr, NI_, 128, ich_W2, ich_b2, 64,  comb_item, 256, 64,  0);
  gemm(stream, agg_ops,   nullptr, NI_, 24,  iop_W1, iop_b1, 128, hidden, 128, 0, 1);
  gemm(stream, hidden,    nullptr, NI_, 128, iop_W2, iop_b2, 64,  comb_item, 256, 128, 0);
  gemm(stream, items,     nullptr, NI_, 16,  ise_W1, ise_b1, 128, hidden, 128, 0, 1);
  gemm(stream, hidden,    nullptr, NI_, 128, ise_W2, ise_b2, 64,  comb_item, 256, 192, 0);
  // item_combined mlp3 on first NI_-1 rows
  gemm(stream, comb_item, nullptr, NI_-1, 256, icb_W1, icb_b1, 128, hidden, 128, 0, 1);
  gemm(stream, hidden,    nullptr, NI_-1, 128, icb_W2, icb_b2, 64,  h2, 64, 0, 1);
  gemm(stream, h2,        nullptr, NI_-1, 64,  icb_W3, icb_b3, 64,  item_out, 64, 0, 0);
  hipMemsetAsync(item_out + (size_t)(NI_-1) * 64, 0, 64 * 4, stream);  // appended zero row

  // ================= Phase B: material embeddings (GAT) =================
  gemm(stream, materials, nullptr, NM_, 8, mat_Wm, nullptr, 64, up_m, 64, 0, 0);
  self_att_kernel<<<blocks1d(NM_,256),256,0,stream>>>(up_m, att_self, NM_, att_all);
  { long n = (long)ENM_ * 32;
    cat32_kernel<<<blocks1d(n,256),256,0,stream>>>(operations, nfm_attr, nfm, ENM_, cat32); }
  gemm(stream, cat32, nullptr, ENM_, 32, mat_Wop, nullptr, 64, opsfeat, 64, 0, 0);
  cross_att_kernel<<<blocks1d(ENM_,256),256,0,stream>>>(up_m, opsfeat, nfm + ENM_,
                                                        att_cross, ENM_, NM_, att_all);
  // global softmax over NM_+ENM_ logits
  const long natt = (long)NM_ + ENM_;
  init_scal_kernel<<<1,32,0,stream>>>(mk, se);
  reduce_max_kernel<<<1024,256,0,stream>>>(att_all, natt, mk);
  reduce_sumexp_kernel<<<1024,256,0,stream>>>(att_all, natt, mk, se);
  hipMemsetAsync(sum_ops, 0, (size_t)NM_ * 64 * 4, stream);
  { long n = (long)ENM_ * 64;
    attn_segsum_kernel<<<blocks1d(n,256),256,0,stream>>>(opsfeat, att_all, nfm + ENM_,
                                                         ENM_, NM_, mk, se, sum_ops); }
  { long n = (long)NM_ * 64;
    mat_emb_kernel<<<blocks1d(n,256),256,0,stream>>>(up_m, sum_ops, att_all, mk, se, NM_, mat_out); }

  // ================= Phase C: operation embeddings =================
  hipMemsetAsync(agg_mat,  0, (size_t)NO_ * 8  * 4, stream);
  hipMemsetAsync(agg_res,  0, (size_t)NO_ * 8  * 4, stream);
  hipMemsetAsync(agg_pred, 0, (size_t)NO_ * 24 * 4, stream);
  hipMemsetAsync(agg_succ, 0, (size_t)NO_ * 24 * 4, stream);
  { long n = (long)ENM_ * 8;
    segsum_kernel<<<blocks1d(n,256),256,0,stream>>>(materials, nfm + ENM_, nfm, ENM_, 8, agg_mat); }
  { long n = (long)ENR_ * 8;
    segsum_kernel<<<blocks1d(n,256),256,0,stream>>>(resources, nfr + ENR_, nfr, ENR_, 8, agg_res); }
  { long n = (long)EP_ * 24;
    segsum_kernel<<<blocks1d(n,256),256,0,stream>>>(operations, prc + EP_, prc, EP_, 24, agg_pred); }
  { long n = (long)EP_ * 24;
    segsum_kernel<<<blocks1d(n,256),256,0,stream>>>(operations, prc, prc + EP_, EP_, 24, agg_succ); }

  // comb_op layout: [pred(0) | succ(64) | res_e(128,8) | mat_e(136,8) | items(144) | self(208)]
  gemm(stream, agg_pred, nullptr, NO_, 24,  opr_W1, opr_b1, 128, hidden, 128, 0, 1);
  gemm(stream, hidden,   nullptr, NO_, 128, opr_W2, opr_b2, 64,  comb_op, 272, 0,   0);
  gemm(stream, agg_succ, nullptr, NO_, 24,  osu_W1, osu_b1, 128, hidden, 128, 0, 1);
  gemm(stream, hidden,   nullptr, NO_, 128, osu_W2, osu_b2, 64,  comb_op, 272, 64,  0);
  gemm(stream, agg_res,  nullptr, NO_, 8,   ore_W1, ore_b1, 128, hidden, 128, 0, 1);
  gemm(stream, hidden,   nullptr, NO_, 128, ore_W2, ore_b2, 8,   comb_op, 272, 128, 0);
  gemm(stream, agg_mat,  nullptr, NO_, 8,   oma_W1, oma_b1, 128, hidden, 128, 0, 1);
  gemm(stream, hidden,   nullptr, NO_, 128, oma_W2, oma_b2, 8,   comb_op, 272, 136, 0);
  gemm(stream, items,    related, NO_, 16,  oit_W1, oit_b1, 128, hidden, 128, 0, 1);
  gemm(stream, hidden,   nullptr, NO_, 128, oit_W2, oit_b2, 64,  comb_op, 272, 144, 0);
  gemm(stream, operations, nullptr, NO_, 24, ose_W1, ose_b1, 128, hidden, 128, 0, 1);
  gemm(stream, hidden,   nullptr, NO_, 128, ose_W2, ose_b2, 64,  comb_op, 272, 208, 0);
  // op_combined mlp3
  gemm(stream, comb_op, nullptr, NO_, 272, ocb_W1, ocb_b1, 128, hidden, 128, 0, 1);
  gemm(stream, hidden,  nullptr, NO_, 128, ocb_W2, ocb_b2, 64,  h2, 64, 0, 1);
  gemm(stream, h2,      nullptr, NO_, 64,  ocb_W3, ocb_b3, 64,  op_out, 64, 0, 0);

  (void)in_sizes; (void)n_in; (void)out_size; (void)ws_size;
}
